// QuantizedLinearWithLora_65343632441947
// MI455X (gfx1250) — matrix operand order repaired and checked by reading, not compiled, not run on hardware
//
#include <hip/hip_runtime.h>
#include <hip/hip_bf16.h>

typedef __attribute__((ext_vector_type(16))) _Float16 v16h;
typedef __attribute__((ext_vector_type(8)))  _Float16 v8h;   // 16B LDS load unit
typedef __attribute__((ext_vector_type(4)))  _Float16 h4;    // 8B LDS store unit
typedef __attribute__((ext_vector_type(8)))  float    v8f;

#define TOKENS 4096
#define IN_F   4096
#define OUT_F  4096
#define RANK   16
#define KCHUNK 64          // == NF4 quant blocksize
#define BM     128         // output tile rows per block
#define BN     64          // output tile cols per block
#define XS_LD  72          // padded LDS stride in halves (144B = 9*16B)
#define SCALING 2.0f       // alpha / rank

static __device__ __forceinline__ v16h join8(v8h lo, v8h hi) {
    return __builtin_shufflevector(lo, hi, 0, 1, 2, 3, 4, 5, 6, 7,
                                           8, 9, 10, 11, 12, 13, 14, 15);
}

// ---------------------------------------------------------------------------
// Kernel 1: XA = x @ A   (4096 x 16), one WMMA tile per wave, K in 32-chunks.
// ---------------------------------------------------------------------------
__global__ __launch_bounds__(256) void lora_xa_wmma(
    const float* __restrict__ x, const float* __restrict__ A,
    float* __restrict__ xa)
{
    const int lane = threadIdx.x & 31;
    const int wave = threadIdx.x >> 5;
    const int m0   = (blockIdx.x * 8 + wave) * 16;

    const int arow = lane & 15;
    const int aofs = (lane >> 4) << 3;   // lanes 16-31: K+8 / K+24 halves
    const int bofs = (lane >> 4) << 3;   // lanes 16-31: K+8 / K+24 rows
    const int bcol = lane & 15;

    v8f acc = {};
    for (int kc = 0; kc < IN_F; kc += 32) {
        const float* xr = x + (size_t)(m0 + arow) * IN_F + kc + aofs;
        const float4 p0 = *(const float4*)(xr);
        const float4 p1 = *(const float4*)(xr + 4);
        const float4 p2 = *(const float4*)(xr + 16);
        const float4 p3 = *(const float4*)(xr + 20);
        v16h af, bf;
        af[0]  = (_Float16)p0.x; af[1]  = (_Float16)p0.y;
        af[2]  = (_Float16)p0.z; af[3]  = (_Float16)p0.w;
        af[4]  = (_Float16)p1.x; af[5]  = (_Float16)p1.y;
        af[6]  = (_Float16)p1.z; af[7]  = (_Float16)p1.w;
        af[8]  = (_Float16)p2.x; af[9]  = (_Float16)p2.y;
        af[10] = (_Float16)p2.z; af[11] = (_Float16)p2.w;
        af[12] = (_Float16)p3.x; af[13] = (_Float16)p3.y;
        af[14] = (_Float16)p3.z; af[15] = (_Float16)p3.w;

        const float* Ar = A + (size_t)(kc + bofs) * RANK + bcol;
#pragma unroll
        for (int i = 0; i < 16; ++i) bf[i] = (_Float16)Ar[(i + (i & 8)) * RANK];

        acc = __builtin_amdgcn_wmma_f32_16x16x32_f16(
            false, af, false, bf, (short)0, acc, false, false);
    }

    const int hi = lane >> 4, col = lane & 15;
#pragma unroll
    for (int j = 0; j < 8; ++j)
        xa[(size_t)(m0 + j + hi * 8) * RANK + col] = acc[j];
}

// ---------------------------------------------------------------------------
// Kernel 2: out = x @ dequant(W)^T + bias + SCALING * XA @ B
// 256 threads = 8 waves; wave (wm,wn) owns a 32x32 subtile = 2x2 WMMA tiles.
// Software-pipelined K loop; LoRA folded in as one augmented K=32 WMMA step.
// ---------------------------------------------------------------------------
__global__ __launch_bounds__(256) void nf4_lora_gemm(
    const float* __restrict__ x, const int* __restrict__ q,
    const float* __restrict__ absmax, const float* __restrict__ cbk,
    const float* __restrict__ bias, const float* __restrict__ Bm,
    const float* __restrict__ xa, float* __restrict__ out)
{
    __shared__ __align__(16) _Float16 xs[BM][XS_LD];  // A-operand tile, f16
    __shared__ __align__(16) _Float16 ws[BN][XS_LD];  // B-operand tile, f16
    __shared__ float cb[16];
    __shared__ float biass[BN];

    const int tid  = threadIdx.x;
    const int lane = tid & 31;
    const int wave = tid >> 5;
    const int wm   = wave & 3;           // 4 M groups of 32 rows
    const int wn   = wave >> 2;          // 2 N groups of 32 cols
    const int m0   = blockIdx.y * BM;
    const int n0   = blockIdx.x * BN;

    if (tid < 16) cb[tid] = cbk[tid];
    if (tid < BN) biass[tid] = bias[n0 + tid];
    __syncthreads();                     // cb[] visible before first dequant

    v8f acc[2][2] = {};

    const int arow = lane & 15;
    const int aofs = (lane >> 4) << 3;
    const int bofs = (lane >> 4) << 3;
    const int bcol = lane & 15;

    // pipeline registers: raw global data for one 64-wide K chunk
    float4 xreg[8];
    int4   qreg[4];
    float  areg[4];

    auto load_chunk = [&](int kc) {
#pragma unroll
        for (int i = 0; i < 8; ++i) {
            const int idx = tid + i * 256;
            xreg[i] = *(const float4*)(x + (size_t)(m0 + (idx >> 4)) * IN_F +
                                       kc + (idx & 15) * 4);
        }
#pragma unroll
        for (int i = 0; i < 4; ++i) {
            const int idx = tid + i * 256;
            qreg[i] = *(const int4*)(q + (size_t)(n0 + (idx >> 4)) * IN_F +
                                     kc + (idx & 15) * 4);
            areg[i] = absmax[(size_t)(n0 + (idx >> 4)) * (IN_F / KCHUNK) +
                             (kc >> 6)];
        }
    };
    auto store_chunk = [&]() {
#pragma unroll
        for (int i = 0; i < 8; ++i) {
            const int idx = tid + i * 256;
            h4 p;
            p[0] = (_Float16)xreg[i].x; p[1] = (_Float16)xreg[i].y;
            p[2] = (_Float16)xreg[i].z; p[3] = (_Float16)xreg[i].w;
            *(h4*)&xs[idx >> 4][(idx & 15) * 4] = p;
        }
#pragma unroll
        for (int i = 0; i < 4; ++i) {
            const int idx = tid + i * 256;
            h4 p;
            p[0] = (_Float16)(cb[qreg[i].x & 15] * areg[i]);
            p[1] = (_Float16)(cb[qreg[i].y & 15] * areg[i]);
            p[2] = (_Float16)(cb[qreg[i].z & 15] * areg[i]);
            p[3] = (_Float16)(cb[qreg[i].w & 15] * areg[i]);
            *(h4*)&ws[idx >> 4][(idx & 15) * 4] = p;
        }
    };
    // one K=32 WMMA step over the staged tiles at half-offset kb
    auto mma_step = [&](int kb) {
        v16h af[2], bf[2];
#pragma unroll
        for (int mi = 0; mi < 2; ++mi) {
            const int rr = wm * 32 + mi * 16 + arow;
            af[mi] = join8(*(const v8h*)&xs[rr][kb + aofs],
                           *(const v8h*)&xs[rr][kb + aofs + 16]);
        }
#pragma unroll
        for (int ni = 0; ni < 2; ++ni) {
            const int cc = wn * 32 + ni * 16 + bcol;
            bf[ni] = join8(*(const v8h*)&ws[cc][kb + bofs],
                           *(const v8h*)&ws[cc][kb + bofs + 16]);
        }
#pragma unroll
        for (int mi = 0; mi < 2; ++mi)
#pragma unroll
            for (int ni = 0; ni < 2; ++ni)
                acc[mi][ni] = __builtin_amdgcn_wmma_f32_16x16x32_f16(
                    false, af[mi], false, bf[ni], (short)0,
                    acc[mi][ni], false, false);
    };

    // ---- main K loop over the base GEMM ----
    load_chunk(0);
    for (int kc = 0; kc < IN_F; kc += KCHUNK) {
        store_chunk();
        __syncthreads();

        if (kc + KCHUNK < IN_F) load_chunk(kc + KCHUNK);   // overlap with WMMA
        if (kc + 2 * KCHUNK < IN_F) {                      // L2 prefetch, d=2
            __builtin_prefetch(x + (size_t)(m0 + (tid >> 1)) * IN_F +
                               kc + 2 * KCHUNK + (tid & 1) * 32, 0, 1);
            if (tid < 128)
                __builtin_prefetch(q + (size_t)(n0 + (tid >> 1)) * IN_F +
                                   kc + 2 * KCHUNK + (tid & 1) * 32, 0, 1);
        }

        mma_step(0);
        mma_step(32);
        __syncthreads();   // compute done before LDS is overwritten
    }

    // ---- LoRA as one augmented K=32 step: A<-XA (pad 16->32), B<-2*B ----
    for (int i = tid; i < BM * 8; i += 256) {     // 128 rows x 8 h4 groups
        const int r = i >> 3, g = i & 7;
        h4 p = {};
        if (g < 4) {
            const float* xr = xa + (size_t)(m0 + r) * RANK + g * 4;
            p[0] = (_Float16)xr[0]; p[1] = (_Float16)xr[1];
            p[2] = (_Float16)xr[2]; p[3] = (_Float16)xr[3];
        }
        *(h4*)&xs[r][g * 4] = p;
    }
    for (int i = tid; i < BN * 8; i += 256) {     // 64 cols x 8 h4 groups
        const int c = i >> 3, g = i & 7;
        h4 p = {};
        if (g < 4) {
#pragma unroll
            for (int j = 0; j < 4; ++j)
                p[j] = (_Float16)(SCALING *
                                  Bm[(size_t)(g * 4 + j) * OUT_F + n0 + c]);
        }
        *(h4*)&ws[c][g * 4] = p;
    }
    __syncthreads();
    mma_step(0);

    // ---- epilogue: + bias, store ----
    const int hi  = lane >> 4;
    const int col = lane & 15;
#pragma unroll
    for (int mi = 0; mi < 2; ++mi)
#pragma unroll
        for (int ni = 0; ni < 2; ++ni) {
            const int nloc = wn * 32 + ni * 16 + col;
#pragma unroll
            for (int j = 0; j < 8; ++j) {
                const int mloc = wm * 32 + mi * 16 + j + hi * 8;
                out[(size_t)(m0 + mloc) * OUT_F + (n0 + nloc)] =
                    acc[mi][ni][j] + biass[nloc];
            }
        }
}

extern "C" void kernel_launch(void* const* d_in, const int* in_sizes, int n_in,
                              void* d_out, int out_size, void* d_ws, size_t ws_size,
                              hipStream_t stream) {
    const float* x      = (const float*)d_in[0];
    const int*   q      = (const int*)d_in[1];
    const float* absmax = (const float*)d_in[2];
    const float* cbk    = (const float*)d_in[3];
    const float* bias   = (const float*)d_in[4];
    const float* A      = (const float*)d_in[5];
    const float* Bm     = (const float*)d_in[6];
    float*       out    = (float*)d_out;
    float*       xa     = (float*)d_ws;      // TOKENS*RANK*4 = 256 KB scratch

    lora_xa_wmma<<<TOKENS / 128, 256, 0, stream>>>(x, A, xa);

    dim3 grid(OUT_F / BN, TOKENS / BM);
    nf4_lora_gemm<<<grid, 256, 0, stream>>>(x, q, absmax, cbk, bias, Bm, xa, out);
}
